// PedestrianTrajectoryModel_87814901334190
// MI455X (gfx1250) — compile-verified, matrix operand order checked by reading, not run on hardware
//
#include <hip/hip_runtime.h>

typedef __attribute__((ext_vector_type(16))) __bf16 v16bf;
typedef __attribute__((ext_vector_type(8)))  float  v8f;

#define NEG_SLOPE 0.2f

// ---- order-preserving float <-> uint keys so segment-max uses HW atomic_max_u32 ----
__device__ __forceinline__ unsigned f2ord(float f) {
    unsigned b = __float_as_uint(f);
    return (b & 0x80000000u) ? ~b : (b | 0x80000000u);
}
__device__ __forceinline__ float ord2f(unsigned u) {
    unsigned b = (u & 0x80000000u) ? (u & 0x7FFFFFFFu) : ~u;
    return __uint_as_float(b);
}

// K0: out = bias (scatter adds accumulate on top), max-key = 0 (below all reals), denom = 0
__global__ void gat_init(float* __restrict__ out, const float* __restrict__ bias,
                         unsigned* __restrict__ maxu, float* __restrict__ denom, int N) {
    int i = blockIdx.x * blockDim.x + threadIdx.x;
    if (i < N * 32) out[i] = bias[i & 31];
    if (i < N) { maxu[i] = 0u; denom[i] = 0.0f; }
}

// K1: h = x @ W via v_wmma_f32_16x16x32_bf16. One wave per 16-node tile; K=32 == F in ONE wmma.
__global__ void gat_gemm_wmma(const float* __restrict__ x, const float* __restrict__ W,
                              float* __restrict__ h, int N) {
    const int lane = threadIdx.x & 31;
    const int wave = threadIdx.x >> 5;
    const int tile = blockIdx.x * (blockDim.x >> 5) + wave;
    const int row0 = tile * 16;
    if (row0 >= N) return;                 // uniform per wave: EXEC stays all-ones for WMMA

    const int m  = lane & 15;              // row (A) / col (B) within tile
    const int kb = (lane >> 4) * 8;        // K-half select per ISA 16-bit A layout

    int row = row0 + m;
    if (row >= N) row = N - 1;             // clamp tail loads; tail stores are guarded

    // A fragment: 16x32 bf16. Lane holds K = kb+{0..7} (vgpr 0-3) and kb+16+{0..7} (vgpr 4-7):
    // two contiguous 8-float runs -> four global_load_b128 per lane.
    const float4* runA0 = (const float4*)(x + (size_t)row * 32 + kb);
    const float4* runA1 = (const float4*)(x + (size_t)row * 32 + kb + 16);
    const float4 a0 = runA0[0], a1 = runA0[1], a2 = runA1[0], a3 = runA1[1];

    v16bf a;
    a[0]  = (__bf16)a0.x; a[1]  = (__bf16)a0.y; a[2]  = (__bf16)a0.z; a[3]  = (__bf16)a0.w;
    a[4]  = (__bf16)a1.x; a[5]  = (__bf16)a1.y; a[6]  = (__bf16)a1.z; a[7]  = (__bf16)a1.w;
    a[8]  = (__bf16)a2.x; a[9]  = (__bf16)a2.y; a[10] = (__bf16)a2.z; a[11] = (__bf16)a2.w;
    a[12] = (__bf16)a3.x; a[13] = (__bf16)a3.y; a[14] = (__bf16)a3.z; a[15] = (__bf16)a3.w;

    const bool full_tile = (row0 + 16 <= N);   // wave-uniform: no exec-mask churn in hot path
    const int  rbase = row0 + ((lane >= 16) ? 8 : 0);

#pragma unroll
    for (int jt = 0; jt < 2; ++jt) {       // two 16-col output tiles (F = 32)
        const int j0 = jt * 16;
        // B fragment: 32x16 bf16, lane holds column j0+m, same K striping as A.
        // Loads are coalesced across lanes (16 consecutive floats per k).
        v16bf b;
#pragma unroll
        for (int v = 0; v < 8; ++v) {
            const int koff = (v < 4) ? (2 * v) : (2 * v + 8);
#pragma unroll
            for (int p = 0; p < 2; ++p) {
                const int k = kb + koff + p;
                b[2 * v + p] = (__bf16)W[k * 32 + j0 + m];
            }
        }
        v8f c = {};
        c = __builtin_amdgcn_wmma_f32_16x16x32_bf16(false, a, false, b,
                                                    (short)0, c, false, false);
        // C/D layout: VGPR r -> row r (lanes 0-15) / row r+8 (lanes 16-31), N = lane&15
        if (full_tile) {
#pragma unroll
            for (int r = 0; r < 8; ++r)
                h[(size_t)(rbase + r) * 32 + j0 + m] = c[r];
        } else {
#pragma unroll
            for (int r = 0; r < 8; ++r) {
                const int rr = rbase + r;
                if (rr < N) h[(size_t)rr * 32 + j0 + m] = c[r];
            }
        }
    }
}

// K2: per-node attention logits, one wave per node (lane = feature), shfl_xor tree reduce
__global__ void gat_logits(const float* __restrict__ h, const float* __restrict__ att_src,
                           const float* __restrict__ att_dst,
                           float* __restrict__ a_src, float* __restrict__ a_dst, int N) {
    const int lane = threadIdx.x & 31;
    const int node = (blockIdx.x * blockDim.x + threadIdx.x) >> 5;
    if (node >= N) return;
    const float v = h[(size_t)node * 32 + lane];
    float s = v * att_src[lane];
    float d = v * att_dst[lane];
#pragma unroll
    for (int o = 16; o > 0; o >>= 1) {
        s += __shfl_xor(s, o, 32);
        d += __shfl_xor(d, o, 32);
    }
    if (lane == 0) { a_src[node] = s; a_dst[node] = d; }
}

// K3: edge logits + segment max (HW u32 atomic max on ordered keys)
__global__ void gat_edge_max(const int* __restrict__ ei, const float* __restrict__ a_src,
                             const float* __restrict__ a_dst, float* __restrict__ e_buf,
                             unsigned* __restrict__ maxu, int E, int N) {
    const long i = (long)blockIdx.x * blockDim.x + threadIdx.x;
    const long M = (long)E + N;
    if (i >= M) return;
    int s, d;
    if (i < E) { s = ei[i]; d = ei[(long)E + i]; } else { s = d = (int)(i - E); }
    float e = a_src[s] + a_dst[d];
    e = (e > 0.0f) ? e : NEG_SLOPE * e;
    e_buf[i] = e;
    atomicMax(&maxu[d], f2ord(e));
}

// K4: ex = exp(e - max[dst]); denom[dst] += ex  (e_buf rewritten in place with ex)
__global__ void gat_edge_expsum(const int* __restrict__ ei, float* __restrict__ e_buf,
                                const unsigned* __restrict__ maxu, float* __restrict__ denom,
                                int E, int N) {
    const long i = (long)blockIdx.x * blockDim.x + threadIdx.x;
    const long M = (long)E + N;
    if (i >= M) return;
    int d;
    if (i < E) { d = ei[(long)E + i]; } else { d = (int)(i - E); }
    const float ex = __expf(e_buf[i] - ord2f(maxu[d]));
    e_buf[i] = ex;
    atomicAdd(&denom[d], ex);
}

// K5: one wave32 per edge, lane = feature (F==32): out[dst,f] += alpha * h[src,f]
__global__ void gat_edge_scatter(const int* __restrict__ ei, const float* __restrict__ e_buf,
                                 const float* __restrict__ denom, const float* __restrict__ h,
                                 float* __restrict__ out, int E, int N) {
    const int  lane = threadIdx.x & 31;
    const long edge = ((long)blockIdx.x * blockDim.x + threadIdx.x) >> 5;
    const long M = (long)E + N;
    if (edge >= M) return;
    int s, d;
    if (edge < E) { s = ei[edge]; d = ei[(long)E + edge]; } else { s = d = (int)(edge - E); }
    const float alpha = e_buf[edge] / denom[d];
    atomicAdd(&out[(long)d * 32 + lane], alpha * h[(long)s * 32 + lane]);
}

extern "C" void kernel_launch(void* const* d_in, const int* in_sizes, int n_in,
                              void* d_out, int out_size, void* d_ws, size_t ws_size,
                              hipStream_t stream) {
    const float* x       = (const float*)d_in[0];
    const int*   ei      = (const int*)  d_in[1];
    const float* W       = (const float*)d_in[2];
    const float* att_src = (const float*)d_in[3];
    const float* att_dst = (const float*)d_in[4];
    const float* bias    = (const float*)d_in[5];

    const int  F = 32;
    const int  N = in_sizes[0] / F;
    const int  E = in_sizes[1] / 2;
    const long M = (long)E + N;
    float* out = (float*)d_out;

    // workspace layout (all L2-resident: ~21 MB total)
    float*    h     = (float*)d_ws;            // N*F
    float*    a_src = h + (size_t)N * F;       // N
    float*    a_dst = a_src + N;               // N
    unsigned* maxu  = (unsigned*)(a_dst + N);  // N
    float*    denom = (float*)(maxu + N);      // N
    float*    e_buf = denom + N;               // E+N

    gat_init<<<(N * F + 255) / 256, 256, 0, stream>>>(out, bias, maxu, denom, N);

    const int tiles = (N + 15) / 16;           // one wave per 16-node tile, 8 waves/block
    gat_gemm_wmma<<<(tiles + 7) / 8, 256, 0, stream>>>(x, W, h, N);

    gat_logits<<<(int)(((long)N * 32 + 255) / 256), 256, 0, stream>>>(h, att_src, att_dst,
                                                                      a_src, a_dst, N);

    gat_edge_max<<<(int)((M + 255) / 256), 256, 0, stream>>>(ei, a_src, a_dst, e_buf, maxu, E, N);
    gat_edge_expsum<<<(int)((M + 255) / 256), 256, 0, stream>>>(ei, e_buf, maxu, denom, E, N);
    gat_edge_scatter<<<(int)((M * 32 + 255) / 256), 256, 0, stream>>>(ei, e_buf, denom, h,
                                                                      out, E, N);
}
